// SelfAttention_13932873908910
// MI455X (gfx1250) — matrix operand order repaired and checked by reading, not compiled, not run on hardware
//
#include <hip/hip_runtime.h>

typedef __bf16 bf16;
typedef __attribute__((ext_vector_type(16))) __bf16 v16bf;
typedef __attribute__((ext_vector_type(8)))  float  v8f;

#define B_   2
#define C_   128
#define G_   32
#define HW_  1024
#define D_   16
#define NSP_ 16384   /* HW_*D_ */
#define EPS_ 1e-5f

/* workspace layout (bytes) */
#define OFF_STATS 0u
#define OFF_QKVW  512u
#define OFF_OUTW  (OFF_QKVW + 384u*128u*2u)
#define OFF_NORMT (OFF_OUTW + 128u*128u*2u)
#define OFF_Q     (OFF_NORMT + (size_t)B_*NSP_*C_*2u)
#define OFF_K     (OFF_Q  + (size_t)B_*D_*HW_*C_*2u)
#define OFF_V     (OFF_K  + (size_t)B_*D_*HW_*C_*2u)
#define OFF_ATTN  (OFF_V  + (size_t)B_*D_*C_*HW_*2u)
#define OFF_OUTT  (OFF_ATTN + (size_t)B_*D_*HW_*HW_*2u)

union BfPack { uint4 u[2]; v16bf v; };

/* ---- gfx1250 async global->LDS path (ASYNCcnt), with portable fallback ---
   Probe-confirmed: builtin exists; param0 is v4i32 in addrspace(1), non-const. */
#if defined(__gfx1250__) && __has_builtin(__builtin_amdgcn_global_load_async_to_lds_b128) && \
    __has_builtin(__builtin_amdgcn_s_wait_asynccnt)
#define ASYNC_LDS 1
#endif

typedef int v4i_gcc __attribute__((vector_size(16)));
typedef __attribute__((address_space(1))) v4i_gcc* gv4i_p;   /* 64-bit global ptr */
typedef __attribute__((address_space(3))) v4i_gcc* lv4i_p;   /* 32-bit LDS offset */

__device__ inline void cp16(void* ldst, const void* gsrc) {
#if ASYNC_LDS
  /* LDS generic address: low 32 bits are the LDS byte offset (ISA 10.2). */
  __builtin_amdgcn_global_load_async_to_lds_b128(
      (gv4i_p)(uintptr_t)gsrc,
      (lv4i_p)(unsigned)(uintptr_t)ldst,
      0, 0);
#else
  *(uint4*)ldst = *(const uint4*)gsrc;
#endif
}
__device__ inline void wait_fills() {
#if ASYNC_LDS
  __builtin_amdgcn_s_wait_asynccnt(0);
#endif
}

/* A operand, 16x32 bf16, row-major [M][K] (ldk = elements per row stride). */
__device__ inline v16bf load_a16x32(const bf16* base, int ldk, int lane) {
  int r = lane & 15, half = lane >> 4;
  const bf16* p = base + (size_t)r * ldk + half * 8;
  BfPack t;
  t.u[0] = *(const uint4*)p;
  t.u[1] = *(const uint4*)(p + 16);
  return t.v;
}

/* B operand, 32x16 bf16, memory stored transposed [N][K]. */
__device__ inline v16bf load_b32x16(const bf16* base, int ldk, int lane) {
  int n = lane & 15, kh = lane >> 4;
  const bf16* p = base + (size_t)n * ldk + kh * 8;
  BfPack t;
  t.u[0] = *(const uint4*)p;
  t.u[1] = *(const uint4*)(p + 16);
  return t.v;
}

#define WMMA_BF16(a,b,c) \
  __builtin_amdgcn_wmma_f32_16x16x32_bf16(false,(a),false,(b),(short)0,(c),false,false)

/* ---------------- 1. GroupNorm statistics: one block per (b,group) -------- */
__global__ void k_gn_stats(const float* __restrict__ x, float* __restrict__ stats) {
  __shared__ float ss[256], ss2[256];
  int bg = blockIdx.x;
  const float* p = x + (size_t)bg * (4 * NSP_);
  float s = 0.f, s2 = 0.f;
  for (int i = threadIdx.x; i < 4 * NSP_; i += 256) { float v = p[i]; s += v; s2 += v * v; }
  ss[threadIdx.x] = s; ss2[threadIdx.x] = s2; __syncthreads();
  for (int o = 128; o > 0; o >>= 1) {
    if ((int)threadIdx.x < o) { ss[threadIdx.x] += ss[threadIdx.x + o]; ss2[threadIdx.x] += ss2[threadIdx.x + o]; }
    __syncthreads();
  }
  if (threadIdx.x == 0) {
    float inv_n = 1.f / (4.f * NSP_);
    float mu = ss[0] * inv_n;
    float var = ss2[0] * inv_n - mu * mu;
    stats[bg * 2] = mu;
    stats[bg * 2 + 1] = rsqrtf(var + EPS_);
  }
}

/* ---------------- 2. weights fp32 -> bf16 -------------------------------- */
__global__ void k_convert_w(const float* __restrict__ qkv_w, const float* __restrict__ out_w,
                            bf16* __restrict__ qkvw_bf, bf16* __restrict__ outw_bf) {
  int i = blockIdx.x * 256 + threadIdx.x;
  if (i < 384 * 128) qkvw_bf[i] = (bf16)qkv_w[i];
  else {
    int j = i - 384 * 128;
    if (j < 128 * 128) outw_bf[j] = (bf16)out_w[j];
  }
}

/* ------- 3. normalize + transpose to [b][n][c] bf16 (B operand layout) ---- */
__global__ void k_normalize(const float* __restrict__ x, const float* __restrict__ gn_w,
                            const float* __restrict__ gn_b, const float* __restrict__ stats,
                            bf16* __restrict__ normT) {
  int idx = blockIdx.x * 256 + threadIdx.x;
  int c = idx & 127;
  int bn = idx >> 7;
  int b = bn >> 14;
  int n = bn & 16383;
  int sg = (b * 32 + (c >> 2)) * 2;
  float mu = stats[sg], rs = stats[sg + 1];
  float v = x[((size_t)(b * 128 + c)) * NSP_ + n];
  normT[idx] = (bf16)(((v - mu) * rs) * gn_w[c] + gn_b[c]);
}

/* ------- 4. QKV GEMM: 8 waves (8 m-tiles) share activation slab in LDS ---- */
__global__ void __launch_bounds__(256) k_qkv_gemm(const bf16* __restrict__ w,
                                                  const bf16* __restrict__ normT,
                                                  bf16* __restrict__ Qb, bf16* __restrict__ Kb,
                                                  bf16* __restrict__ Vb) {
  __shared__ __align__(16) bf16 bs[2][64 * 32];     /* [n][32c] slab, 4 KB x2 */
  int tid = threadIdx.x, lane = tid & 31, wv = tid >> 5;
  int bi = blockIdx.x;
  int b   = bi / (3 * 256);
  int mg  = (bi / 256) % 3;
  int n_t = bi % 256;
  int m0 = (mg * 8 + wv) * 16, n0 = n_t * 64;

  const bf16* bsrc = normT + ((size_t)(b * NSP_ + n0)) * 128;
  auto fill = [&](int buf, int kc) {
    int s = tid;                       /* row n = s>>2, seg = s&3 */
    int n = s >> 2, seg = s & 3;
    cp16(&bs[buf][n * 32 + seg * 8], bsrc + (size_t)n * 128 + kc * 32 + seg * 8);
  };

  v8f acc[4] = {};
  fill(0, 0);
#pragma unroll
  for (int kc = 0; kc < 4; ++kc) {
    int cur = kc & 1;
    wait_fills();
    __syncthreads();
    if (kc + 1 < 4) fill(1 - cur, kc + 1);
    v16bf a = load_a16x32(w + (size_t)m0 * 128 + kc * 32, 128, lane);
#pragma unroll
    for (int j = 0; j < 4; ++j) {
      v16bf bb = load_b32x16(&bs[cur][j * 16 * 32], 32, lane);
      acc[j] = WMMA_BF16(a, bb, acc[j]);
    }
    __syncthreads();
  }

  int half = lane >> 4, ln = lane & 15;
#pragma unroll
  for (int j = 0; j < 4; ++j) {
#pragma unroll
    for (int i = 0; i < 8; ++i) {
      int o = m0 + i + half * 8;
      int n = n0 + j * 16 + ln;
      int d = n & 15, hw = n >> 4;
      bf16 val = (bf16)acc[j][i];
      if (o < 128)       Qb[((size_t)((b * 16 + d) * 1024 + hw)) * 128 + o] = val;
      else if (o < 256)  Kb[((size_t)((b * 16 + d) * 1024 + hw)) * 128 + (o - 128)] = val;
      else               Vb[((size_t)((b * 16 + d) * 128 + (o - 256))) * 1024 + hw] = val;
    }
  }
}

/* ------- 5. scores: 8 waves share the Q tile (all 16 d) staged in LDS ----- */
__global__ void __launch_bounds__(256) k_scores(const bf16* __restrict__ Qb,
                                                const bf16* __restrict__ Kb,
                                                bf16* __restrict__ attn) {
  __shared__ __align__(16) bf16 qs[2][256 * 32];    /* [(d*16+hw)][32c], 16 KB x2 */
  int tid = threadIdx.x, lane = tid & 31, wv = tid >> 5;
  int bi = blockIdx.x;
  int b    = bi / (64 * 8);
  int hw_t = (bi / 8) % 64;
  int yxg  = bi % 8;
  int hw0 = hw_t * 16;
  int yx0 = (yxg * 8 + wv) * 16;

  auto fill = [&](int buf, int kc) {
#pragma unroll
    for (int q = 0; q < 4; ++q) {
      int s = q * 256 + tid;
      int row = s >> 2, seg = s & 3;     /* row = d*16 + hwl */
      int d = row >> 4, hwl = row & 15;
      cp16(&qs[buf][row * 32 + seg * 8],
           Qb + ((size_t)((b * 16 + d) * 1024 + hw0 + hwl)) * 128 + kc * 32 + seg * 8);
    }
  };

  v8f acc[16] = {};
  fill(0, 0);
#pragma unroll
  for (int kc = 0; kc < 4; ++kc) {
    int cur = kc & 1;
    wait_fills();
    __syncthreads();
    if (kc + 1 < 4) fill(1 - cur, kc + 1);
#pragma unroll
    for (int d = 0; d < 16; ++d) {
      v16bf a  = load_a16x32(&qs[cur][d * 16 * 32], 32, lane);
      v16bf bb = load_b32x16(Kb + ((size_t)((b * 16 + d) * 1024 + yx0)) * 128 + kc * 32, 128, lane);
      acc[d] = WMMA_BF16(a, bb, acc[d]);
    }
    __syncthreads();
  }

  const float rscale = 0.08838834764831845f;   /* 1/sqrt(C=128) */
  int half = lane >> 4, ln = lane & 15;
#pragma unroll
  for (int i = 0; i < 8; ++i) {
    float m = -3.4e38f;
#pragma unroll
    for (int d = 0; d < 16; ++d) m = fmaxf(m, acc[d][i]);
    float e[16]; float s = 0.f;
#pragma unroll
    for (int d = 0; d < 16; ++d) { e[d] = __expf((acc[d][i] - m) * rscale); s += e[d]; }
    float inv = 1.f / s;
    int hw = hw0 + i + half * 8, yx = yx0 + ln;
#pragma unroll
    for (int d = 0; d < 16; ++d)
      attn[((size_t)((b * 16 + d) * 1024 + hw)) * 1024 + yx] = (bf16)(e[d] * inv);
  }
}

/* ------- 6. attn x V per (b,d): 8 waves share the V k-slab in LDS --------- */
__global__ void __launch_bounds__(256) k_attn_out(const bf16* __restrict__ attn,
                                                  const bf16* __restrict__ Vb,
                                                  bf16* __restrict__ outT) {
  __shared__ __align__(16) bf16 vs[2][128 * 32];    /* [c][32 yx], 8 KB x2 */
  int tid = threadIdx.x, lane = tid & 31, wv = tid >> 5;
  int bi = blockIdx.x;
  int b   = bi / (16 * 8);
  int d   = (bi / 8) % 16;
  int hwg = bi % 8;
  int hw0 = (hwg * 8 + wv) * 16;

  const bf16* vsrc = Vb + ((size_t)((b * 16 + d) * 128)) * 1024;
  auto fill = [&](int buf, int kc) {
#pragma unroll
    for (int q = 0; q < 2; ++q) {
      int s = q * 256 + tid;
      int c = s >> 2, seg = s & 3;
      cp16(&vs[buf][c * 32 + seg * 8], vsrc + (size_t)c * 1024 + kc * 32 + seg * 8);
    }
  };

  v8f acc[8] = {};
  fill(0, 0);
  for (int kc = 0; kc < 32; ++kc) {
    int cur = kc & 1;
    wait_fills();
    __syncthreads();
    if (kc + 1 < 32) fill(1 - cur, kc + 1);
    v16bf a = load_a16x32(attn + ((size_t)((b * 16 + d) * 1024 + hw0)) * 1024 + kc * 32, 1024, lane);
#pragma unroll
    for (int j = 0; j < 8; ++j) {
      v16bf bb = load_b32x16(&vs[cur][j * 16 * 32], 32, lane);
      acc[j] = WMMA_BF16(a, bb, acc[j]);
    }
    __syncthreads();
  }

  int half = lane >> 4, ln = lane & 15;
#pragma unroll
  for (int j = 0; j < 8; ++j)
#pragma unroll
    for (int i = 0; i < 8; ++i) {
      int hw = hw0 + i + half * 8, c = j * 16 + ln;
      outT[((size_t)(b * NSP_ + hw * 16 + d)) * 128 + c] = (bf16)acc[j][i];
    }
}

/* ------- 7. projection + bias + residual: 8 waves cover full M=128 -------- */
__global__ void __launch_bounds__(256) k_final(const bf16* __restrict__ w,
                                               const bf16* __restrict__ outT,
                                               const float* __restrict__ bias,
                                               const float* __restrict__ x,
                                               float* __restrict__ out) {
  __shared__ __align__(16) bf16 bs[2][64 * 32];     /* 4 KB x2 */
  int tid = threadIdx.x, lane = tid & 31, wv = tid >> 5;
  int bi = blockIdx.x;
  int b   = bi / 256;
  int n_t = bi % 256;
  int m0 = wv * 16, n0 = n_t * 64;

  const bf16* bsrc = outT + ((size_t)(b * NSP_ + n0)) * 128;
  auto fill = [&](int buf, int kc) {
    int s = tid;
    int n = s >> 2, seg = s & 3;
    cp16(&bs[buf][n * 32 + seg * 8], bsrc + (size_t)n * 128 + kc * 32 + seg * 8);
  };

  v8f acc[4] = {};
  fill(0, 0);
#pragma unroll
  for (int kc = 0; kc < 4; ++kc) {
    int cur = kc & 1;
    wait_fills();
    __syncthreads();
    if (kc + 1 < 4) fill(1 - cur, kc + 1);
    v16bf a = load_a16x32(w + (size_t)m0 * 128 + kc * 32, 128, lane);
#pragma unroll
    for (int j = 0; j < 4; ++j) {
      v16bf bb = load_b32x16(&bs[cur][j * 16 * 32], 32, lane);
      acc[j] = WMMA_BF16(a, bb, acc[j]);
    }
    __syncthreads();
  }

  int half = lane >> 4, ln = lane & 15;
#pragma unroll
  for (int j = 0; j < 4; ++j)
#pragma unroll
    for (int i = 0; i < 8; ++i) {
      int o = m0 + i + half * 8;
      int n = n0 + j * 16 + ln;
      size_t idx = ((size_t)(b * 128 + o)) * NSP_ + n;
      out[idx] = acc[j][i] + bias[o] + x[idx];
    }
}

extern "C" void kernel_launch(void* const* d_in, const int* in_sizes, int n_in,
                              void* d_out, int out_size, void* d_ws, size_t ws_size,
                              hipStream_t stream) {
  (void)in_sizes; (void)n_in; (void)out_size; (void)ws_size;
  const float* x     = (const float*)d_in[0];
  const float* gn_w  = (const float*)d_in[1];
  const float* gn_b  = (const float*)d_in[2];
  const float* qkv_w = (const float*)d_in[3];
  const float* out_w = (const float*)d_in[4];
  const float* out_b = (const float*)d_in[5];
  float* out = (float*)d_out;

  char* ws = (char*)d_ws;
  float* stats   = (float*)(ws + OFF_STATS);
  bf16*  qkvw_bf = (bf16*)(ws + OFF_QKVW);
  bf16*  outw_bf = (bf16*)(ws + OFF_OUTW);
  bf16*  normT   = (bf16*)(ws + OFF_NORMT);
  bf16*  Qb      = (bf16*)(ws + OFF_Q);
  bf16*  Kb      = (bf16*)(ws + OFF_K);
  bf16*  Vb      = (bf16*)(ws + OFF_V);
  bf16*  attn    = (bf16*)(ws + OFF_ATTN);
  bf16*  outT    = (bf16*)(ws + OFF_OUTT);

  k_gn_stats <<<B_ * G_,        256, 0, stream>>>(x, stats);
  k_convert_w<<<(384*128 + 128*128) / 256, 256, 0, stream>>>(qkv_w, out_w, qkvw_bf, outw_bf);
  k_normalize<<<(B_ * NSP_ * C_) / 256, 256, 0, stream>>>(x, gn_w, gn_b, stats, normT);
  k_qkv_gemm <<<B_ * 3 * 256,  256, 0, stream>>>(qkvw_bf, normT, Qb, Kb, Vb);
  k_scores   <<<B_ * 64 * 8,   256, 0, stream>>>(Qb, Kb, attn);
  k_attn_out <<<B_ * 16 * 8,   256, 0, stream>>>(attn, Vb, outT);
  k_final    <<<B_ * 256,      256, 0, stream>>>(outw_bf, outT, out_b, x, out);
}
